// ExphormerAttention_75453985456957
// MI455X (gfx1250) — matrix operand order repaired and checked by reading, not compiled, not run on hardware
//
#include <hip/hip_runtime.h>
#include <hip/hip_bf16.h>

#define N_NODES   50000
#define NUM_EDGES 1600000
#define IN_DIM    64
#define OUT_DIM   64
#define NUM_HEADS 4
#define HEAD_DIM  16   // one WMMA column tile per head

typedef _Float16 v16h __attribute__((ext_vector_type(16)));
typedef float    v8f  __attribute__((ext_vector_type(8)));

// ---------------------------------------------------------------------------
// A-matrix loader: 16 rows x 32 K of a row-major f32 [*,64] matrix, converted
// to f16 in the CDNA5 16-bit A layout (16x32 MxK):
//   lanes 0-15 : row = lane,     halves 0-7 = K 0-7,   halves 8-15 = K 16-23
//   lanes16-31 : row = lane-16,  halves 0-7 = K 8-15,  halves 8-15 = K 24-31
// kstep selects K base 0 or 32.
// ---------------------------------------------------------------------------
__device__ inline v16h load_A_f16(const float* __restrict__ tile, int lane, int kstep) {
  const int row = lane & 15;
  const int kb  = ((lane & 16) ? 8 : 0) + kstep * 32;
  const float* p = tile + row * IN_DIM + kb;
  v16h a;
#pragma unroll
  for (int i = 0; i < 8; ++i) {
    a[i]     = (_Float16)p[i];        // K = kb + i
    a[8 + i] = (_Float16)p[16 + i];   // K = kb + 16 + i
  }
  return a;
}

// ---------------------------------------------------------------------------
// B-matrix loader for Y = X * W^T with W stored row-major [OUT,IN]:
// B = W^T (K x N), column n of B = row n of W (contiguous 64 floats).
// CDNA5 16-bit B layout (32x16 KxN):
//   lanes 0-15 : col = n0+lane,    halves 0-7 = K 0-7,   halves 8-15 = K 16-23
//   lanes16-31 : col = n0+lane-16, halves 0-7 = K 8-15,  halves 8-15 = K 24-31
// ---------------------------------------------------------------------------
__device__ inline v16h load_B_f16(const float* __restrict__ W, int n0, int lane, int kstep) {
  const int col = n0 + (lane & 15);
  const int kb  = ((lane & 16) ? 8 : 0) + kstep * 32;
  const float* p = W + col * IN_DIM + kb;
  v16h b;
#pragma unroll
  for (int i = 0; i < 8; ++i) {
    b[i]     = (_Float16)p[i];
    b[8 + i] = (_Float16)p[16 + i];
  }
  return b;
}

// 16x64 output tile: acc[t] is the 16x16 f32 C tile for columns [16t,16t+16).
__device__ inline void gemm_16x64(const float* __restrict__ Arows,
                                  const float* __restrict__ W,
                                  const float* __restrict__ bias,
                                  int lane, v8f acc[4]) {
  const v16h a0 = load_A_f16(Arows, lane, 0);
  const v16h a1 = load_A_f16(Arows, lane, 1);
#pragma unroll
  for (int t = 0; t < 4; ++t) {
    const v16h b0 = load_B_f16(W, t * 16, lane, 0);
    const v16h b1 = load_B_f16(W, t * 16, lane, 1);
    v8f c = {};
    c = __builtin_amdgcn_wmma_f32_16x16x32_f16(false, a0, false, b0, (short)0, c, false, false);
    c = __builtin_amdgcn_wmma_f32_16x16x32_f16(false, a1, false, b1, (short)0, c, false, false);
    const float bb = bias[t * 16 + (lane & 15)];   // C layout: col = lane&15 in every VGPR
#pragma unroll
    for (int r = 0; r < 8; ++r) c[r] += bb;
    acc[t] = c;
  }
}

// ---------------------------------------------------------------------------
// Kernel 0: zero the output accumulator (atomics add into it every launch).
// ---------------------------------------------------------------------------
__global__ __launch_bounds__(256)
void exph_zero_kernel(float* __restrict__ p, int n) {
  int i = blockIdx.x * 256 + threadIdx.x;
  if (i < n) p[i] = 0.0f;
}

// ---------------------------------------------------------------------------
// Kernel 1: node projections Q/K/V. blockIdx.y selects the projection,
// each wave computes one 16-node x 64-feature tile (8 WMMAs).
// ---------------------------------------------------------------------------
__global__ __launch_bounds__(128)
void exph_proj_kernel(const float* __restrict__ x,
                      const float* __restrict__ Wq, const float* __restrict__ bq,
                      const float* __restrict__ Wk, const float* __restrict__ bk,
                      const float* __restrict__ Wv, const float* __restrict__ bv,
                      float* __restrict__ Q, float* __restrict__ K, float* __restrict__ V) {
  const int wave = threadIdx.x >> 5;
  const int lane = threadIdx.x & 31;
  const int tile = blockIdx.x * 4 + wave;
  if (tile >= N_NODES / 16) return;          // wave-uniform: EXEC stays all-1 for WMMA
  const int row0 = tile * 16;

  const float *W, *b; float* O;
  if (blockIdx.y == 0)      { W = Wq; b = bq; O = Q; }
  else if (blockIdx.y == 1) { W = Wk; b = bk; O = K; }
  else                      { W = Wv; b = bv; O = V; }

  v8f acc[4];
  gemm_16x64(x + row0 * IN_DIM, W, b, lane, acc);

  const int rowoff = (lane & 16) ? 8 : 0;
  const int col    = lane & 15;
#pragma unroll
  for (int t = 0; t < 4; ++t)
#pragma unroll
    for (int r = 0; r < 8; ++r)
      O[(row0 + rowoff + r) * OUT_DIM + t * 16 + col] = acc[t][r];
}

// ---------------------------------------------------------------------------
// Kernel 2: fused edge pipeline. Per wave: 16 edges.
//   1) Ee tile via WMMA (never written to memory),
//   2) per-head score reduction via 16-lane xor shuffles,
//   3) exp(clip) and atomic scatter of V[src]*score into out[dst].
// Head h == accumulator tile h (D == 16 == WMMA N-tile).
// ---------------------------------------------------------------------------
__global__ __launch_bounds__(128)
void exph_edge_kernel(const float* __restrict__ edge_attr,
                      const int*   __restrict__ edge_index,
                      const float* __restrict__ We, const float* __restrict__ be,
                      const float* __restrict__ Q,
                      const float* __restrict__ K,
                      const float* __restrict__ V,
                      float* __restrict__ out) {
  const int wave = threadIdx.x >> 5;
  const int lane = threadIdx.x & 31;
  const int e0   = (blockIdx.x * 4 + wave) * 16;

  // Prefetch the next wave-tile of the edge_attr stream into cache.
  {
    const int pe = (e0 + 64 < NUM_EDGES) ? (e0 + 64) : 0;
    __builtin_prefetch(edge_attr + (size_t)pe * IN_DIM + lane * 16, 0, 1);
  }

  v8f acc[4];                                   // Ee[16 edges][4 heads][16 d]
  gemm_16x64(edge_attr + (size_t)e0 * IN_DIM, We, be, lane, acc);

  const int col = lane & 15;                    // d within head
  const int mo  = (lane & 16) ? 8 : 0;          // edge offset of this lane-half
  const int* __restrict__ srcp = edge_index;
  const int* __restrict__ dstp = edge_index + NUM_EDGES;

#pragma unroll
  for (int r = 0; r < 8; ++r) {
    const int e = e0 + mo + r;                  // this lane-half's edge for C-register r
    const int s = srcp[e];
    const int d = dstp[e];
    const float* __restrict__ Krow = K + (size_t)s * OUT_DIM;
    const float* __restrict__ Qrow = Q + (size_t)d * OUT_DIM;
    const float* __restrict__ Vrow = V + (size_t)s * OUT_DIM;
    float* __restrict__ Orow = out + (size_t)d * OUT_DIM;

#pragma unroll
    for (int h = 0; h < 4; ++h) {
      // part = K[s][h][col] * Q[d][h][col] * Ee[e][h][col]
      float part = Krow[h * 16 + col] * Qrow[h * 16 + col] * acc[h][r];
      // sum over d = 0..15 (xor masks < 16 stay within each 16-lane half)
      part += __shfl_xor(part, 1);
      part += __shfl_xor(part, 2);
      part += __shfl_xor(part, 4);
      part += __shfl_xor(part, 8);
      float sc = part * 0.25f;                  // 1/sqrt(D), D = 16
      sc = fminf(5.0f, fmaxf(-5.0f, sc));
      const float w = __expf(sc);
      // each lane owns one (edge, head, d) output element
      atomicAdd(&Orow[h * 16 + col], Vrow[h * 16 + col] * w);
    }
  }
}

extern "C" void kernel_launch(void* const* d_in, const int* in_sizes, int n_in,
                              void* d_out, int out_size, void* d_ws, size_t ws_size,
                              hipStream_t stream) {
  const float* x         = (const float*)d_in[0];
  const float* edge_attr = (const float*)d_in[1];
  const int*   edge_idx  = (const int*)  d_in[2];
  const float* Wq = (const float*)d_in[3];
  const float* bq = (const float*)d_in[4];
  const float* Wk = (const float*)d_in[5];
  const float* bk = (const float*)d_in[6];
  const float* We = (const float*)d_in[7];
  const float* be = (const float*)d_in[8];
  const float* Wv = (const float*)d_in[9];
  const float* bv = (const float*)d_in[10];
  float* out = (float*)d_out;

  // Workspace: Q, K, V projections (3 * N * 64 f32 = 38.4 MB)
  float* Q = (float*)d_ws;
  float* K = Q + (size_t)N_NODES * OUT_DIM;
  float* V = K + (size_t)N_NODES * OUT_DIM;

  // 1) zero output (atomic accumulation target)
  {
    const int n = N_NODES * OUT_DIM;
    exph_zero_kernel<<<(n + 255) / 256, 256, 0, stream>>>(out, n);
  }
  // 2) Q/K/V projections: 3125 tiles of 16 nodes, 4 waves per block
  {
    dim3 grid((N_NODES / 16 + 3) / 4, 3);
    exph_proj_kernel<<<grid, 128, 0, stream>>>(x, Wq, bq, Wk, bk, Wv, bv, Q, K, V);
  }
  // 3) fused edge kernel: 100000 tiles of 16 edges, 4 waves per block
  {
    dim3 grid(NUM_EDGES / (16 * 4));
    exph_edge_kernel<<<grid, 128, 0, stream>>>(edge_attr, edge_idx, We, be, Q, K, V, out);
  }
}
